// Int8LoRALinear_33809982554585
// MI455X (gfx1250) — matrix operand order repaired and checked by reading, not compiled, not run on hardware
//
#include <hip/hip_runtime.h>

typedef __attribute__((ext_vector_type(16))) _Float16 v16h;
typedef __attribute__((ext_vector_type(8)))  _Float16 v8h;
typedef __attribute__((ext_vector_type(8)))  float    v8f;
typedef __attribute__((ext_vector_type(8)))  int      v8i;
typedef __attribute__((ext_vector_type(4)))  int      v4i;

#define DIN   4096
#define DOUT  11008
#define MTOT  8192          // B * S = 4 * 2048
#define LORA_SCALE 1.0f     // ALPHA / RANK = 16/16

// CDNA5 async global->LDS path (guarded; falls back to reg staging if builtins absent)
#if defined(__gfx1250__) && \
    __has_builtin(__builtin_amdgcn_global_load_async_to_lds_b128) && \
    __has_builtin(__builtin_amdgcn_s_wait_asynccnt)
#define USE_ASYNC 1
#else
#define USE_ASYNC 0
#endif

// ---------------- conversion kernels ----------------

__global__ void cvt_x_kernel(const float* __restrict__ x, _Float16* __restrict__ x16, int n8) {
    int t = blockIdx.x * blockDim.x + threadIdx.x;
    if (t >= n8) return;
    size_t base = (size_t)t * 8;
    v8f f = *(const v8f*)(x + base);
    v8h h;
#pragma unroll
    for (int i = 0; i < 8; i++) h[i] = (_Float16)f[i];
    *(v8h*)(x16 + base) = h;
}

__global__ void cvt_w_kernel(const int* __restrict__ w, _Float16* __restrict__ w16, int n8) {
    int t = blockIdx.x * blockDim.x + threadIdx.x;
    if (t >= n8) return;
    size_t base = (size_t)t * 8;
    v8i wi = *(const v8i*)(w + base);
    v8h h;
#pragma unroll
    for (int i = 0; i < 8; i++) h[i] = (_Float16)(float)wi[i];
    *(v8h*)(w16 + base) = h;
}

// lora_b [DOUT,16] f32 -> [DOUT,32] f16, cols 16..31 zero
__global__ void pack_lb_kernel(const float* __restrict__ lb, _Float16* __restrict__ lb16, int dout) {
    int o = blockIdx.x * blockDim.x + threadIdx.x;
    if (o >= dout) return;
    v8h zero = {};
#pragma unroll
    for (int g = 0; g < 2; g++) {
        v8f f = *(const v8f*)(lb + (size_t)o * 16 + g * 8);
        v8h h;
#pragma unroll
        for (int i = 0; i < 8; i++) h[i] = (_Float16)(f[i] * LORA_SCALE);
        *(v8h*)(lb16 + (size_t)o * 32 + g * 8) = h;
    }
    *(v8h*)(lb16 + (size_t)o * 32 + 16) = zero;
    *(v8h*)(lb16 + (size_t)o * 32 + 24) = zero;
}

// xa[m, r] = sum_k x[m,k] * lora_a[r,k]   -> f16 [MTOT, 32], cols 16..31 zero
__global__ __launch_bounds__(256) void xa_kernel(const float* __restrict__ x,
                                                 const float* __restrict__ la,
                                                 _Float16* __restrict__ xa16) {
    __shared__ float red[256][16];
    const int m = blockIdx.x;
    const int tid = threadIdx.x;
    float acc[16] = {};
    const float* xr = x + (size_t)m * DIN;
    for (int k = tid; k < DIN; k += 256) {
        float xv = xr[k];
#pragma unroll
        for (int r = 0; r < 16; r++) acc[r] += xv * la[r * DIN + k];
    }
#pragma unroll
    for (int r = 0; r < 16; r++) red[tid][r] = acc[r];
    __syncthreads();
    for (int s = 128; s > 0; s >>= 1) {
        if (tid < s) {
#pragma unroll
            for (int r = 0; r < 16; r++) red[tid][r] += red[tid + s][r];
        }
        __syncthreads();
    }
    if (tid < 16)       xa16[(size_t)m * 32 + tid] = (_Float16)red[0][tid];
    else if (tid < 32)  xa16[(size_t)m * 32 + tid] = (_Float16)0.0f;
}

// ---------------- main GEMM ----------------
// out[m, o] = (sum_k x16[m,k]*w16[o,k]) * scale[o] + sum_r xa16[m,r]*lb16[o,r]

union Frag { v16h v; v8h h[2]; };

#define BM 128
#define BN 256
#define BK 64
#define LDT 72   // padded LDS row stride in halves (144 B, keeps 16B alignment)

#if USE_ASYNC
#define NBUF 2
typedef __attribute__((address_space(1))) v4i gas_v4i;   // global (AS1) int4
typedef __attribute__((address_space(3))) v4i las_v4i;   // LDS (AS3) int4

__device__ __forceinline__ void async_cp16(const _Float16* g, _Float16* l) {
    // generic->AS1: identical numeric address; generic->AS3: LDS offset is the low 32 bits
    gas_v4i* gp = (gas_v4i*)(unsigned long long)(size_t)g;
    las_v4i* lp = (las_v4i*)(unsigned int)(size_t)l;
    __builtin_amdgcn_global_load_async_to_lds_b128(gp, lp, 0, 0);
}
#else
#define NBUF 1
#endif

__device__ __forceinline__ void compute_tile(const _Float16* Asb, const _Float16* Bsb,
                                             v8f (&c)[4][4], int wm, int wn, int lr, int hi) {
#pragma unroll
    for (int kk = 0; kk < BK; kk += 32) {
        Frag a[4], b[4];
#pragma unroll
        for (int mt = 0; mt < 4; mt++) {
            int row = wm + mt * 16 + lr;
            // ISA A 16x32 f16 layout: halves 0..7 -> K=kk+8*hi.., halves 8..15 -> K=kk+16+8*hi..
            a[mt].h[0] = *(const v8h*)(Asb + row * LDT + kk + 8 * hi);
            a[mt].h[1] = *(const v8h*)(Asb + row * LDT + kk + 16 + 8 * hi);
        }
#pragma unroll
        for (int nt = 0; nt < 4; nt++) {
            int row = wn + nt * 16 + lr;
            // ISA B 32x16 f16 layout: lane = N, halves 0..7 -> K=kk+8*hi.., halves 8..15 -> K=kk+16+8*hi..
            b[nt].h[0] = *(const v8h*)(Bsb + row * LDT + kk + 8 * hi);
            b[nt].h[1] = *(const v8h*)(Bsb + row * LDT + kk + 16 + 8 * hi);
        }
#pragma unroll
        for (int mt = 0; mt < 4; mt++)
#pragma unroll
            for (int nt = 0; nt < 4; nt++)
                c[mt][nt] = __builtin_amdgcn_wmma_f32_16x16x32_f16(
                    false, a[mt].v, false, b[nt].v, (short)0, c[mt][nt], false, false);
    }
}

__global__ __launch_bounds__(256) void gemm_kernel(const _Float16* __restrict__ x16,
                                                   const _Float16* __restrict__ w16,
                                                   const float*    __restrict__ scale,
                                                   const _Float16* __restrict__ xa16,
                                                   const _Float16* __restrict__ lb16,
                                                   float* __restrict__ out) {
    __shared__ __align__(16) _Float16 As[NBUF][BM * LDT];
    __shared__ __align__(16) _Float16 Bs[NBUF][BN * LDT];

    const int tid  = threadIdx.x;
    const int m0   = blockIdx.x * BM;
    const int n0   = blockIdx.y * BN;
    const int wid  = tid >> 5;
    const int lane = tid & 31;
    const int lr   = lane & 15;
    const int hi   = lane >> 4;
    const int wm   = (wid & 1) * 64;   // 2 waves along M
    const int wn   = (wid >> 1) * 64;  // 4 waves along N

    v8f c[4][4] = {};

#if USE_ASYNC
    // -------- double-buffered pipeline with GLOBAL_LOAD_ASYNC_TO_LDS_B128 --------
    // per wave per tile: 12 async ops (4 for A, 8 for B)
    auto issue = [&](int buf, int k0) {
#pragma unroll
        for (int i = 0; i < 4; i++) {
            int ch = tid + i * 256, row = ch >> 3, cc = (ch & 7) * 8;
            async_cp16(x16 + (size_t)(m0 + row) * DIN + k0 + cc, &As[buf][row * LDT + cc]);
        }
#pragma unroll
        for (int i = 0; i < 8; i++) {
            int ch = tid + i * 256, row = ch >> 3, cc = (ch & 7) * 8;
            async_cp16(w16 + (size_t)(n0 + row) * DIN + k0 + cc, &Bs[buf][row * LDT + cc]);
        }
    };
    issue(0, 0);
    int cur = 0;
    for (int k0 = 0; k0 < DIN; k0 += BK) {
        if (k0 + BK < DIN) {
            issue(cur ^ 1, k0 + BK);                 // prefetch next tile into other buffer
            __builtin_amdgcn_s_wait_asynccnt(12);    // drain current buffer's 12, keep next 12 in flight
        } else {
            __builtin_amdgcn_s_wait_asynccnt(0);
        }
        __syncthreads();                             // all waves' current-buffer loads visible
        compute_tile(As[cur], Bs[cur], c, wm, wn, lr, hi);
        __syncthreads();                             // all waves done reading before reuse of buffer
        cur ^= 1;
    }
#else
    // -------- fallback: register staging + single LDS buffer --------
    for (int k0 = 0; k0 < DIN; k0 += BK) {
        v8h ar[4], br[8];
#pragma unroll
        for (int i = 0; i < 4; i++) {
            int ch = tid + i * 256, row = ch >> 3, cc = (ch & 7) * 8;
            ar[i] = *(const v8h*)(x16 + (size_t)(m0 + row) * DIN + k0 + cc);
        }
#pragma unroll
        for (int i = 0; i < 8; i++) {
            int ch = tid + i * 256, row = ch >> 3, cc = (ch & 7) * 8;
            br[i] = *(const v8h*)(w16 + (size_t)(n0 + row) * DIN + k0 + cc);
        }
        __syncthreads();
#pragma unroll
        for (int i = 0; i < 4; i++) {
            int ch = tid + i * 256, row = ch >> 3, cc = (ch & 7) * 8;
            *(v8h*)(&As[0][row * LDT + cc]) = ar[i];
        }
#pragma unroll
        for (int i = 0; i < 8; i++) {
            int ch = tid + i * 256, row = ch >> 3, cc = (ch & 7) * 8;
            *(v8h*)(&Bs[0][row * LDT + cc]) = br[i];
        }
        __syncthreads();
        compute_tile(As[0], Bs[0], c, wm, wn, lr, hi);
    }
#endif

    // ---- epilogue: scale, then LoRA correction as one extra WMMA per tile ----
    Frag al[4], bl[4];
#pragma unroll
    for (int mt = 0; mt < 4; mt++) {
        size_t m = (size_t)(m0 + wm + mt * 16 + lr);
        al[mt].h[0] = *(const v8h*)(xa16 + m * 32 + 8 * hi);
        al[mt].h[1] = *(const v8h*)(xa16 + m * 32 + 16 + 8 * hi);   // zero pad region
    }
#pragma unroll
    for (int nt = 0; nt < 4; nt++) {
        size_t o = (size_t)(n0 + wn + nt * 16 + lr);
        bl[nt].h[0] = *(const v8h*)(lb16 + o * 32 + 8 * hi);
        bl[nt].h[1] = *(const v8h*)(lb16 + o * 32 + 16 + 8 * hi);   // zero pad region
    }
    float sc[4];
#pragma unroll
    for (int nt = 0; nt < 4; nt++) sc[nt] = scale[n0 + wn + nt * 16 + lr];

#pragma unroll
    for (int mt = 0; mt < 4; mt++) {
#pragma unroll
        for (int nt = 0; nt < 4; nt++) {
            v8f cs = c[mt][nt] * sc[nt];
            v8f d  = __builtin_amdgcn_wmma_f32_16x16x32_f16(
                false, al[mt].v, false, bl[nt].v, (short)0, cs, false, false);
            int ncol = n0 + wn + nt * 16 + lr;
#pragma unroll
            for (int v = 0; v < 8; v++) {
                // C/D layout: VGPR v -> M = 8*hi + v, N = lr
                out[(size_t)(m0 + wm + mt * 16 + 8 * hi + v) * DOUT + ncol] = d[v];
            }
        }
    }
}

// ---------------- launcher ----------------

extern "C" void kernel_launch(void* const* d_in, const int* in_sizes, int n_in,
                              void* d_out, int out_size, void* d_ws, size_t ws_size,
                              hipStream_t stream) {
    const float* x      = (const float*)d_in[0];
    const int*   w_i8   = (const int*)  d_in[1];
    const float* wscale = (const float*)d_in[2];
    const float* lora_a = (const float*)d_in[3];
    const float* lora_b = (const float*)d_in[4];
    float* out = (float*)d_out;

    // workspace layout (bytes)
    const size_t W16_BYTES = (size_t)DOUT * DIN * 2;   //  90,177,536
    const size_t X16_BYTES = (size_t)MTOT * DIN * 2;   //  67,108,864
    const size_t XA_BYTES  = (size_t)MTOT * 32 * 2;    //     524,288
    char* ws = (char*)d_ws;
    _Float16* w16  = (_Float16*)ws;
    _Float16* x16  = (_Float16*)(ws + W16_BYTES);
    _Float16* xa16 = (_Float16*)(ws + W16_BYTES + X16_BYTES);
    _Float16* lb16 = (_Float16*)(ws + W16_BYTES + X16_BYTES + XA_BYTES);

    // 1) x f32 -> f16
    {
        int n8 = (MTOT * DIN) / 8;
        cvt_x_kernel<<<(n8 + 255) / 256, 256, 0, stream>>>(x, x16, n8);
    }
    // 2) weights int32 -> f16 (exact for int8 range)
    {
        int n8 = (DOUT * DIN) / 8;
        cvt_w_kernel<<<(n8 + 255) / 256, 256, 0, stream>>>(w_i8, w16, n8);
    }
    // 3) pack lora_b into zero-padded f16 [DOUT,32]
    pack_lb_kernel<<<(DOUT + 255) / 256, 256, 0, stream>>>(lora_b, lb16, DOUT);
    // 4) xa = x @ A^T, f16 zero-padded [MTOT,32]
    xa_kernel<<<MTOT, 256, 0, stream>>>(x, lora_a, xa16);
    // 5) WMMA GEMM + scale + LoRA epilogue
    {
        dim3 grid(MTOT / BM, DOUT / BN);               // 64 x 43
        gemm_kernel<<<grid, 256, 0, stream>>>(x16, w16, wscale, xa16, lb16, out);
    }
}
